// ExpertParallelWrapper_68496138436742
// MI455X (gfx1250) — matrix operand order repaired and checked by reading, not compiled, not run on hardware
//
#include <hip/hip_runtime.h>
#include <hip/hip_bf16.h>
#include <stdint.h>

// Problem constants (from reference): B=2, S=2048, H=1024, F=4096, E=8
#define T_TOK 4096   // B*S tokens
#define H_DIM 1024
#define F_DIM 4096
#define N_EXP 8

#define USE_ASYNC_LDS 1

typedef __attribute__((ext_vector_type(16))) __bf16 v16bf;
typedef __attribute__((ext_vector_type(8)))  float  v8f;

__device__ __forceinline__ __bf16 f2bf_rne(float f) {
    union { float f; uint32_t u; } x; x.f = f;
    uint32_t r = x.u + 0x7FFFu + ((x.u >> 16) & 1u);
    union { unsigned short s; __bf16 b; } y;
    y.s = (unsigned short)(r >> 16);
    return y.b;
}

// truncating f32 -> bf16 store: lowers to ds_store_b16_d16_hi (zero extra VALU)
__device__ __forceinline__ void st_bf16_hi(__bf16* p, float f) {
    union { float f; uint32_t u; } x; x.f = f;
    *(unsigned short*)p = (unsigned short)(x.u >> 16);
}

__device__ __forceinline__ void wait_async0() {
#if USE_ASYNC_LDS
#if __has_builtin(__builtin_amdgcn_s_wait_asynccnt)
    __builtin_amdgcn_s_wait_asynccnt(0);
#else
    asm volatile("s_wait_asynccnt 0" ::: "memory");
#endif
#endif
}

// ---------------------------------------------------------------------------
// 0) zero the per-expert token counters
// ---------------------------------------------------------------------------
__global__ void moe_init_counts(int* cnt) {
    if (threadIdx.x < N_EXP) cnt[threadIdx.x] = 0;
}

// ---------------------------------------------------------------------------
// 1) convert x (fp32) -> xb (bf16), vectorized 4 elements / thread
// ---------------------------------------------------------------------------
__global__ __launch_bounds__(256)
void moe_cvt_x(const float* __restrict__ x, __bf16* __restrict__ xb) {
    int i = blockIdx.x * blockDim.x + threadIdx.x;          // 0 .. T*H/4-1
    const float4 v = ((const float4*)x)[i];
    union { __bf16 b[4]; uint2 q; } o;
    o.b[0] = f2bf_rne(v.x); o.b[1] = f2bf_rne(v.y);
    o.b[2] = f2bf_rne(v.z); o.b[3] = f2bf_rne(v.w);
    ((uint2*)xb)[i] = o.q;
}

// ---------------------------------------------------------------------------
// 2) router: one wave32 per token -> top-1 expert + softmax weight,
//    append token into its expert's dispatch list.
// ---------------------------------------------------------------------------
__global__ __launch_bounds__(128)
void moe_router(const float* __restrict__ x, const float* __restrict__ wr,
                float* __restrict__ wgt, int* __restrict__ cnt,
                int* __restrict__ perm) {
    const int lane = threadIdx.x & 31;
    const int wid  = threadIdx.x >> 5;
    const int t    = blockIdx.x * 4 + wid;

    const float* xr = x + (size_t)t * H_DIM;
    float acc[N_EXP];
#pragma unroll
    for (int e = 0; e < N_EXP; ++e) acc[e] = 0.0f;

    for (int k = lane; k < H_DIM; k += 32) {
        float xv = xr[k];
        const float4* w = (const float4*)(wr + (size_t)k * N_EXP);
        float4 w0 = w[0], w1 = w[1];
        acc[0] += xv * w0.x; acc[1] += xv * w0.y;
        acc[2] += xv * w0.z; acc[3] += xv * w0.w;
        acc[4] += xv * w1.x; acc[5] += xv * w1.y;
        acc[6] += xv * w1.z; acc[7] += xv * w1.w;
    }
#pragma unroll
    for (int off = 16; off > 0; off >>= 1)
#pragma unroll
        for (int e = 0; e < N_EXP; ++e)
            acc[e] += __shfl_xor(acc[e], off, 32);

    if (lane == 0) {
        float m = acc[0]; int s = 0;
#pragma unroll
        for (int e = 1; e < N_EXP; ++e)
            if (acc[e] > m) { m = acc[e]; s = e; }          // first-max = argmax
        float denom = 0.0f;
#pragma unroll
        for (int e = 0; e < N_EXP; ++e) denom += __expf(acc[e] - m);
        wgt[t] = 1.0f / denom;
        int pos = atomicAdd(&cnt[s], 1);
        perm[s * T_TOK + pos] = t;
    }
}

// ---------------------------------------------------------------------------
// 3/4) gathered expert GEMM, 128 threads = 4 waves.
//   Block tile 64(M)x64(N); each wave owns 32x32 -> 4 WMMAs / K-step with
//   fragment reuse. K-step = 32 (bf16). LDS double-buffered, 1 barrier/step.
//   A (bf16, token-gathered rows) is DMA'd global->LDS via async-load;
//   B (fp32 row-major KxN) loaded as float4, truncated to bf16, stored
//   K-transposed in LDS ([n][k]) so fragments are contiguous b128 ds loads.
// ---------------------------------------------------------------------------
template<int KDIM, int NDIM, bool STAGE2>
__global__ __launch_bounds__(128)
void moe_expert_gemm(const __bf16* __restrict__ Abase,
                     const float*  __restrict__ Wbase,
                     const float*  __restrict__ bias,
                     const int*    __restrict__ cnt,
                     const int*    __restrict__ perm,
                     const float*  __restrict__ wgt,
                     __bf16* __restrict__ hout,
                     float*  __restrict__ yout) {
    const int e = blockIdx.z;
    const int c = cnt[e];
    const int mbase = blockIdx.y * 64;
    if (mbase >= c) return;                     // uniform exit (EXEC safe)
    const int nbase = blockIdx.x * 64;

    __shared__ __bf16 As[2][64][32];            // [buf][m][k]
    __shared__ __bf16 Bs[2][64][32];            // [buf][n][k] (K-transposed)

    const int tid  = threadIdx.x;
    const int lane = tid & 31;
    const int wid  = tid >> 5;
    const int wm   = (wid >> 1) * 32;           // wave tile origin in block
    const int wn   = (wid & 1) * 32;

    // --- A tile cooperative copy: 64 rows x two 16-elem (32B) halves ------
    const int arow = tid >> 1;                  // 0..63
    const int alc  = (tid & 1) * 16;            // 0 / 16
    int am = mbase + arow;
    if (am >= c) am = c - 1;                    // clamp: junk rows discarded later
    const int atok = perm[e * T_TOK + am];
    const __bf16* Arow = Abase + (size_t)atok * KDIM + alc;

    // --- B tile: 32 k-rows x 4 segments of 16 fp32 ------------------------
    const int bkr = tid >> 2;                   // 0..31 (k within tile)
    const int bnc = (tid & 3) * 16;             // 0,16,32,48 (n within tile)
    const float* W = Wbase + (size_t)e * KDIM * NDIM + nbase + bnc;

    v8f acc[2][2] = {};
    float4 bq[4];

    auto issue_a = [&](int k0, int buf) {
#if USE_ASYNC_LDS
        uint32_t l0 = (uint32_t)(uintptr_t)&As[buf][arow][alc];
        const void* g0 = (const void*)(Arow + k0);
        asm volatile(
            "global_load_async_to_lds_b128 %0, %1, off\n\t"
            "global_load_async_to_lds_b128 %0, %1, off offset:16"
            :: "v"(l0), "v"(g0) : "memory");
#else
        uint4 q0 = *(const uint4*)(Arow + k0);
        uint4 q1 = *(const uint4*)(Arow + k0 + 8);
        *(uint4*)(&As[buf][arow][alc])     = q0;
        *(uint4*)(&As[buf][arow][alc + 8]) = q1;
#endif
    };
    auto load_b = [&](int k0) {
        const float4* wr = (const float4*)(W + (size_t)(k0 + bkr) * NDIM);
        bq[0] = wr[0]; bq[1] = wr[1]; bq[2] = wr[2]; bq[3] = wr[3];
    };
    auto store_b = [&](int buf) {
#pragma unroll
        for (int q = 0; q < 4; ++q) {
            st_bf16_hi(&Bs[buf][bnc + q * 4 + 0][bkr], bq[q].x);
            st_bf16_hi(&Bs[buf][bnc + q * 4 + 1][bkr], bq[q].y);
            st_bf16_hi(&Bs[buf][bnc + q * 4 + 2][bkr], bq[q].z);
            st_bf16_hi(&Bs[buf][bnc + q * 4 + 3][bkr], bq[q].w);
        }
    };

    // prologue: stage K-tile 0 into buffer 0
    issue_a(0, 0);
    load_b(0);
    store_b(0);

    const int hs = lane >> 4;                   // lane half select
    const int rl = lane & 15;

    int buf = 0;
    for (int k0 = 0; k0 < KDIM; k0 += 32) {
        wait_async0();                          // my async copies into `buf` done
        __syncthreads();                        // everyone's tile `buf` visible

        const bool more = (k0 + 32) < KDIM;
        if (more) {                             // stage next tile into buf^1
            issue_a(k0 + 32, buf ^ 1);          // (safe: buf^1 readers passed barrier)
            load_b(k0 + 32);
        }

        // fragments (ISA 16-bit A/B VGPR layouts), b128 LDS loads
        union { v16bf v; uint4 q[2]; } fa[2], fb[2];
#pragma unroll
        for (int mi = 0; mi < 2; ++mi) {
            const __bf16* row = &As[buf][wm + mi * 16 + rl][0];
            fa[mi].q[0] = *(const uint4*)(row + hs * 8);
            fa[mi].q[1] = *(const uint4*)(row + 16 + hs * 8);
        }
#pragma unroll
        for (int ni = 0; ni < 2; ++ni) {
            const __bf16* rowb = &Bs[buf][wn + ni * 16 + rl][0];
            fb[ni].q[0] = *(const uint4*)(rowb + hs * 8);
            fb[ni].q[1] = *(const uint4*)(rowb + 16 + hs * 8);
        }
#pragma unroll
        for (int mi = 0; mi < 2; ++mi)
#pragma unroll
            for (int ni = 0; ni < 2; ++ni)
                acc[mi][ni] = __builtin_amdgcn_wmma_f32_16x16x32_bf16(
                    false, fa[mi].v, false, fb[ni].v, (short)0,
                    acc[mi][ni], false, false);

        if (more) store_b(buf ^ 1);
        buf ^= 1;
    }

    // ---- epilogue: C/D layout: lane = col, VGPR r -> row hs*8 + r --------
#pragma unroll
    for (int mi = 0; mi < 2; ++mi) {
#pragma unroll
        for (int r = 0; r < 8; ++r) {
            const int m = mbase + wm + mi * 16 + hs * 8 + r;
            if (m < c) {
                const int tok = perm[e * T_TOK + m];
                const float tw = STAGE2 ? wgt[tok] : 0.0f;
#pragma unroll
                for (int ni = 0; ni < 2; ++ni) {
                    const int col = nbase + wn + ni * 16 + rl;
                    float v = acc[mi][ni][r] + bias[(size_t)e * NDIM + col];
                    if (!STAGE2) {
                        v = fmaxf(v, 0.0f);     // ReLU
                        hout[(size_t)tok * NDIM + col] = f2bf_rne(v);
                    } else {
                        yout[(size_t)tok * NDIM + col] = v * tw;
                    }
                }
            }
        }
    }
}

// ---------------------------------------------------------------------------
// launch
// ---------------------------------------------------------------------------
extern "C" void kernel_launch(void* const* d_in, const int* in_sizes, int n_in,
                              void* d_out, int out_size, void* d_ws, size_t ws_size,
                              hipStream_t stream) {
    const float* x  = (const float*)d_in[0];   // [T, H]
    const float* wr = (const float*)d_in[1];   // [H, E]
    const float* w1 = (const float*)d_in[2];   // [E, H, F]
    const float* b1 = (const float*)d_in[3];   // [E, F]
    const float* w2 = (const float*)d_in[4];   // [E, F, H]
    const float* b2 = (const float*)d_in[5];   // [E, H]
    float* out = (float*)d_out;                // [T, H]

    char* ws = (char*)d_ws;
    __bf16* xb   = (__bf16*)(ws);                                   //  8 MiB
    __bf16* hbuf = (__bf16*)(ws + (size_t)T_TOK * H_DIM * 2);       // 32 MiB
    float*  wgt  = (float*) (ws + (size_t)T_TOK * (H_DIM + F_DIM) * 2);
    int*    perm = (int*)   ((char*)wgt + (size_t)T_TOK * 4);
    int*    cnt  = (int*)   ((char*)perm + (size_t)N_EXP * T_TOK * 4);

    moe_init_counts<<<1, 32, 0, stream>>>(cnt);

    moe_cvt_x<<<(T_TOK * H_DIM / 4) / 256, 256, 0, stream>>>(x, xb);

    moe_router<<<T_TOK / 4, 128, 0, stream>>>(x, wr, wgt, cnt, perm);

    // GEMM1: [tokens_e, H] @ [H, F] -> hbuf  (grid covers worst case M = T)
    dim3 g1(F_DIM / 64, T_TOK / 64, N_EXP);
    moe_expert_gemm<H_DIM, F_DIM, false><<<g1, 128, 0, stream>>>(
        xb, w1, b1, cnt, perm, wgt, hbuf, nullptr);

    // GEMM2: [tokens_e, F] @ [F, H] -> out (scaled by routing weight)
    dim3 g2(H_DIM / 64, T_TOK / 64, N_EXP);
    moe_expert_gemm<F_DIM, H_DIM, true><<<g2, 128, 0, stream>>>(
        hbuf, w2, b2, cnt, perm, wgt, nullptr, out);
}
